// Attention_17291538334224
// MI455X (gfx1250) — matrix operand order repaired and checked by reading, not compiled, not run on hardware
//
#include <hip/hip_runtime.h>

typedef __bf16 v16bf __attribute__((ext_vector_type(16)));
typedef float  v8f   __attribute__((ext_vector_type(8)));
typedef __bf16 v2bf  __attribute__((ext_vector_type(2)));
typedef float  v2f   __attribute__((ext_vector_type(2)));

#define B_      4
#define S_      4096
#define D_      256
#define TQ      64            // q rows per block
#define TK      32            // kv rows per tile
#define NW      (TQ / 16)     // 4 waves
#define NTHR    (NW * 32)     // 128 threads
#define NEGINF  (-1e30f)
#define QSCALE  (0.0625f * 1.44269504088896340736f)  // 1/sqrt(256) * log2(e)

#define KSTRIDE 132           // words per kv row of K in LDS (128 data + 4 pad)
#define QSTRIDE 132           // words per q row of Q in LDS
#define VSTRIDE 20            // words per d row of V^T in LDS (16 pairs + 4 pad)

union Frag {
    v16bf        bf;
    unsigned int u[8];
    uint4        q[2];
};

__device__ __forceinline__ unsigned int pk2bf(float a, float b) {
    v2f f; f[0] = a; f[1] = b;
    v2bf h = __builtin_convertvector(f, v2bf);   // v_cvt_pk_bf16_f32
    unsigned int r;
    __builtin_memcpy(&r, &h, 4);
    return r;
}

// bytes 0..rlim set to 0x01 (clamped), others 0
__device__ __forceinline__ unsigned long long causal_bytes(int rlim) {
    if (rlim >= 7) return 0x0101010101010101ull;
    if (rlim < 0)  return 0ull;
    return 0x0101010101010101ull >> (8 * (7 - rlim));
}

__global__ __launch_bounds__(NTHR, 1)
void fattn_cdna5(const float* __restrict__ Qg, const float* __restrict__ Kg,
                 const float* __restrict__ Vg, const unsigned char* __restrict__ Mg,
                 float* __restrict__ Og)
{
    __shared__ unsigned int Qpk[TQ * QSTRIDE];   // Q tile (scaled), row-major bf16 pairs
    __shared__ unsigned int Kpk[TK * KSTRIDE];   // K tile, row-major bf16 pairs
    __shared__ unsigned int Vpk[D_ * VSTRIDE];   // V tile transposed, bf16 kv-pairs

    const int tid  = threadIdx.x;
    const int lane = tid & 31;
    const int wid  = tid >> 5;
    const int ln16 = lane & 15;
    const int g    = lane >> 4;          // half-wave (lane group)

    const int blocksPerBatch = S_ / TQ;  // 64
    const int b    = blockIdx.x / blocksPerBatch;
    const int qb   = (blockIdx.x % blocksPerBatch) * TQ;
    const int q0   = qb + wid * 16;      // wave's first q row
    const int qrow = q0 + ln16;          // this lane's q row (softmax state row)

    const float* Kb = Kg + (size_t)b * S_ * D_;
    const float* Vb = Vg + (size_t)b * S_ * D_;
    const unsigned char* Mb = Mg + (size_t)b * S_;

    // ---- stage Q tile once: fp32 -> (bf16 * QSCALE), row-major pairs ----
    {
        const int row = tid >> 1;                 // 0..63
        const int seg = (tid & 1) * 128;          // 128 floats per thread
        const float* src = Qg + ((size_t)b * S_ + qb + row) * D_ + seg;
        unsigned int* dst = &Qpk[row * QSTRIDE + (seg >> 1)];
#pragma unroll
        for (int i = 0; i < 16; ++i) {
            float4 v0 = *(const float4*)(src + 8 * i);
            float4 v1 = *(const float4*)(src + 8 * i + 4);
            uint4 w;
            w.x = pk2bf(v0.x * QSCALE, v0.y * QSCALE);
            w.y = pk2bf(v0.z * QSCALE, v0.w * QSCALE);
            w.z = pk2bf(v1.x * QSCALE, v1.y * QSCALE);
            w.w = pk2bf(v1.z * QSCALE, v1.w * QSCALE);
            *(uint4*)(dst + 4 * i) = w;
        }
    }

    v8f o[16];
#pragma unroll
    for (int n = 0; n < 16; ++n)
#pragma unroll
        for (int r = 0; r < 8; ++r) o[n][r] = 0.0f;
    float mrow = NEGINF;
    float lrow = 0.0f;

    const int ntiles = qb / TK + (TQ / TK);      // block-uniform causal tile count
    for (int ti = 0; ti < ntiles; ++ti) {
        const int tb = ti * TK;

        __syncthreads();                          // previous tile's LDS reads done
        // ---- stage K tile: 32 rows x 256 d, bf16 pairs, b128 stores ----
        {
            const int kv  = tid >> 2;             // 0..31
            const int seg = (tid & 3) * 64;       // 64 floats per thread
            const float* src = Kb + (size_t)(tb + kv) * D_ + seg;
            unsigned int* dst = &Kpk[kv * KSTRIDE + (seg >> 1)];
#pragma unroll
            for (int i = 0; i < 8; ++i) {
                float4 v0 = *(const float4*)(src + 8 * i);
                float4 v1 = *(const float4*)(src + 8 * i + 4);
                uint4 w;
                w.x = pk2bf(v0.x, v0.y);
                w.y = pk2bf(v0.z, v0.w);
                w.z = pk2bf(v1.x, v1.y);
                w.w = pk2bf(v1.z, v1.w);
                *(uint4*)(dst + 4 * i) = w;
            }
        }
        // ---- stage V tile transposed: Vpk[d][kvpair], staggered stores ----
        {
            const int p  = tid & 15;              // kv pair 0..15
            const int gg = tid >> 4;              // 0..7 -> 32 d each
            const int dbase = gg * 32;
            const float* r0 = Vb + (size_t)(tb + 2 * p) * D_ + dbase;
            const float* r1 = r0 + D_;
#pragma unroll
            for (int i = 0; i < 8; ++i) {
                const int ii = (i + gg) & 7;      // stagger across half-waves
                float4 va = *(const float4*)(r0 + 4 * ii);
                float4 vb = *(const float4*)(r1 + 4 * ii);
                unsigned int* dst = &Vpk[(dbase + 4 * ii) * VSTRIDE + p];
                dst[0 * VSTRIDE] = pk2bf(va.x, vb.x);
                dst[1 * VSTRIDE] = pk2bf(va.y, vb.y);
                dst[2 * VSTRIDE] = pk2bf(va.z, vb.z);
                dst[3 * VSTRIDE] = pk2bf(va.w, vb.w);
            }
        }
        // ---- prefetch next tile's K/V segments toward the WGP ----
        if (ti + 1 < ntiles) {
            const float* nk = Kb + (size_t)(tb + TK + (tid >> 2)) * D_ + (tid & 3) * 64;
            __builtin_prefetch(nk, 0, 0);
            __builtin_prefetch(nk + 32, 0, 0);
            const float* nv = Vb + (size_t)(tb + TK + 2 * (tid & 15)) * D_ + (tid >> 4) * 32;
            __builtin_prefetch(nv, 0, 0);
            __builtin_prefetch(nv + D_, 0, 0);
        }
        __syncthreads();

        if (tb <= q0 + 15) {                      // wave-uniform causal skip
            // ---- S^T = K_sub x Q^T : two 16x16 f32 tiles over 8 d-chunks ----
            v8f st0, st1;
#pragma unroll
            for (int r = 0; r < 8; ++r) { st0[r] = 0.0f; st1[r] = 0.0f; }
#pragma unroll
            for (int c = 0; c < 8; ++c) {
                Frag qf, ka, kb2;
                const unsigned int* qp = &Qpk[(wid * 16 + ln16) * QSTRIDE + c * 16 + g * 4];
                qf.q[0] = *(const uint4*)qp;
                qf.q[1] = *(const uint4*)(qp + 8);
                const unsigned int* k0 = &Kpk[ln16 * KSTRIDE + c * 16 + g * 4];
                const unsigned int* k1 = &Kpk[(16 + ln16) * KSTRIDE + c * 16 + g * 4];
                ka.q[0]  = *(const uint4*)k0;
                ka.q[1]  = *(const uint4*)(k0 + 8);
                kb2.q[0] = *(const uint4*)k1;
                kb2.q[1] = *(const uint4*)(k1 + 8);
                st0 = __builtin_amdgcn_wmma_f32_16x16x32_bf16(false, ka.bf,  false, qf.bf, (short)0, st0, false, false);
                st1 = __builtin_amdgcn_wmma_f32_16x16x32_bf16(false, kb2.bf, false, qf.bf, (short)0, st1, false, false);
            }

            // ---- merged causal+padding byte masks, applied in place ----
            unsigned long long mb0 = *(const unsigned long long*)(Mb + tb + g * 8);
            unsigned long long mb1 = *(const unsigned long long*)(Mb + tb + 16 + g * 8);
            mb0 &= causal_bytes(qrow - (tb + g * 8));
            mb1 &= causal_bytes(qrow - (tb + 16 + g * 8));
            float rmax = NEGINF;
#pragma unroll
            for (int r = 0; r < 8; ++r) {
                st0[r] = ((mb0 >> (8 * r)) & 1ull) ? st0[r] : NEGINF;
                rmax = fmaxf(rmax, st0[r]);
                st1[r] = ((mb1 >> (8 * r)) & 1ull) ? st1[r] : NEGINF;
                rmax = fmaxf(rmax, st1[r]);
            }
            rmax = fmaxf(rmax, __shfl_xor(rmax, 16, 32));       // combine half-waves (same q)
            const float mnew = fmaxf(mrow, rmax);
            float lsum = 0.0f;
            Frag pa;                                            // P in A-layout, no shuffles needed
#pragma unroll
            for (int r = 0; r < 8; r += 2) {
                float pa0 = __builtin_amdgcn_exp2f(st0[r] - mnew);
                float pa1 = __builtin_amdgcn_exp2f(st0[r + 1] - mnew);
                float pb0 = __builtin_amdgcn_exp2f(st1[r] - mnew);
                float pb1 = __builtin_amdgcn_exp2f(st1[r + 1] - mnew);
                lsum += pa0 + pa1 + pb0 + pb1;
                pa.u[r / 2]     = pk2bf(pa0, pa1);
                pa.u[4 + r / 2] = pk2bf(pb0, pb1);
            }
            lsum += __shfl_xor(lsum, 16, 32);

            // alpha == 1.0 exactly when no row max increased -> update is exact either way
            const float alpha = __builtin_amdgcn_exp2f(mrow - mnew);
            lrow = lrow * alpha + lsum;
            mrow = mnew;

            // ---- rescale O only if some row's max actually increased ----
            if (__ballot(alpha < 1.0f) != 0ull) {
                float osc[8];
#pragma unroll
                for (int r = 0; r < 8; ++r) osc[r] = __shfl(alpha, g * 8 + r, 32);
#pragma unroll
                for (int n = 0; n < 16; ++n)
#pragma unroll
                    for (int r = 0; r < 8; ++r) o[n][r] *= osc[r];
            }

            // ---- O += P x V : 16 output d-chunks ----
#pragma unroll
            for (int n = 0; n < 16; ++n) {
                Frag vf;
                const unsigned int* vp = &Vpk[(n * 16 + ln16) * VSTRIDE + g * 4];
                vf.q[0] = *(const uint4*)vp;
                vf.q[1] = *(const uint4*)(vp + 8);
                o[n] = __builtin_amdgcn_wmma_f32_16x16x32_bf16(false, pa.bf, false, vf.bf, (short)0, o[n], false, false);
            }
        }
    }

    // ---- epilogue: divide by row sums (broadcast into C-layout), store fp32 ----
    float linv[8];
#pragma unroll
    for (int r = 0; r < 8; ++r) {
        float lr = __shfl(lrow, g * 8 + r, 32);
        linv[r] = 1.0f / lr;
    }
    float* Ob = Og + (size_t)b * S_ * D_;
#pragma unroll
    for (int n = 0; n < 16; ++n)
#pragma unroll
        for (int r = 0; r < 8; ++r)
            Ob[(size_t)(q0 + g * 8 + r) * D_ + n * 16 + ln16] = o[n][r] * linv[r];
}

extern "C" void kernel_launch(void* const* d_in, const int* in_sizes, int n_in,
                              void* d_out, int out_size, void* d_ws, size_t ws_size,
                              hipStream_t stream) {
    (void)in_sizes; (void)n_in; (void)out_size; (void)d_ws; (void)ws_size;
    const float* Q = (const float*)d_in[0];
    const float* K = (const float*)d_in[1];
    const float* V = (const float*)d_in[2];
    const unsigned char* M = (const unsigned char*)d_in[3];
    float* O = (float*)d_out;
    dim3 grid(B_ * (S_ / TQ));   // 256 blocks
    dim3 block(NTHR);            // 128 threads = 4 waves
    hipLaunchKernelGGL(fattn_cdna5, grid, block, 0, stream, Q, K, V, M, O);
}
